// Allegro_54674933678510
// MI455X (gfx1250) — matrix operand order repaired and checked by reading, not compiled, not run on hardware
//
#include <hip/hip_runtime.h>
#include <math.h>

// ---------------------------------------------------------------------------
// Types for WMMA fragments (gfx1250, wave32)
// ---------------------------------------------------------------------------
typedef __attribute__((ext_vector_type(8)))  float          v8f;
typedef __attribute__((ext_vector_type(16))) __bf16         v16bf;
typedef __attribute__((ext_vector_type(4)))  unsigned int   u32x4;
typedef __attribute__((ext_vector_type(8)))  unsigned int   u32x8;

#define EPSILON     0.2425356250363330f   // 1/sqrt(17)
#define SQRT3       1.7320508075688772f
#define INV_SQRT3   0.5773502691896258f
#define INV_SQRT2   0.7071067811865475f

__device__ __forceinline__ unsigned short f2bf(float f) {
    unsigned int u = __builtin_bit_cast(unsigned int, f);
    u += 0x7FFFu + ((u >> 16) & 1u);   // round-to-nearest-even
    return (unsigned short)(u >> 16);
}

// A fragment: 16x32 bf16 (M x K).  lanes 0-15: row M=lane, elems 0-15 = K0-15 ;
// lanes 16-31: same rows, K16-31.
__device__ __forceinline__ v16bf load_a_frag(const unsigned short* __restrict__ Arow,
                                             int kb, int laneHi) {
    const u32x4 lo = *reinterpret_cast<const u32x4*>(Arow + kb + 16 * laneHi);
    const u32x4 hi = *reinterpret_cast<const u32x4*>(Arow + kb + 16 * laneHi + 8);
    u32x8 c;
    c[0] = lo[0]; c[1] = lo[1]; c[2] = lo[2]; c[3] = lo[3];
    c[4] = hi[0]; c[5] = hi[1]; c[6] = hi[2]; c[7] = hi[3];
    return __builtin_bit_cast(v16bf, c);
}

__device__ __forceinline__ v16bf load_b_frag(const unsigned short* __restrict__ p) {
    u32x8 raw = *reinterpret_cast<const u32x8*>(p);
    return __builtin_bit_cast(v16bf, raw);
}

__device__ __forceinline__ v8f wmma_bf16(v16bf a, v16bf b, v8f c) {
    return __builtin_amdgcn_wmma_f32_16x16x32_bf16(false, a, false, b,
                                                   (short)0, c, false, false);
}

// ---------------------------------------------------------------------------
// Weight packing: W (K x N, f32 row-major) -> bf16 fragment-major.
// Fragment (kt,nt): 32 lanes x 16 bf16; lane holds B[kt*32+16*laneHi+i][nt*16+laneLo].
// ---------------------------------------------------------------------------
__global__ __launch_bounds__(256) void pack_weights(const float* __restrict__ W,
                                                    unsigned short* __restrict__ out,
                                                    int K, int N) {
    int tid = blockIdx.x * 256 + threadIdx.x;
    int colTiles = N >> 4;
    int total = (K >> 5) * colTiles * 32;
    if (tid >= total) return;
    int lane = tid & 31;
    int frag = tid >> 5;
    int kt = frag / colTiles, nt = frag % colTiles;
    int laneLo = lane & 15, laneHi = lane >> 4;
    int kbase = kt * 32 + 16 * laneHi;
    int col = nt * 16 + laneLo;
    unsigned short* dst = out + (long)frag * 512 + lane * 16;
    #pragma unroll
    for (int i = 0; i < 16; ++i)
        dst[i] = f2bf(W[(long)(kbase + i) * N + col]);
}

// ---------------------------------------------------------------------------
// x (E x 512 f32) -> bf16 columns [0,512) of xcb (stride 640)
// ---------------------------------------------------------------------------
typedef __attribute__((ext_vector_type(4))) unsigned short u16x4;
__global__ __launch_bounds__(256) void xcopy(const float* __restrict__ x,
                                             unsigned short* __restrict__ xcb,
                                             long n4) {
    long tid = (long)blockIdx.x * 256 + threadIdx.x;
    if (tid >= n4) return;
    long e = tid >> 7;            // 128 float4-groups per row
    int  f = (int)(tid & 127) * 4;
    const float4 v = *reinterpret_cast<const float4*>(x + e * 512 + f);
    u16x4 o;
    o[0] = f2bf(v.x); o[1] = f2bf(v.y); o[2] = f2bf(v.z); o[3] = f2bf(v.w);
    *reinterpret_cast<u16x4*>(xcb + e * 640 + f) = o;
}

// ---------------------------------------------------------------------------
// Per-edge geometry: Y = [1, sqrt3 * nhat], envelope(r)
// ---------------------------------------------------------------------------
__global__ __launch_bounds__(256) void geom(const float* __restrict__ vectors,
                                            float* __restrict__ Yv,
                                            float* __restrict__ env, long E) {
    long e = (long)blockIdx.x * 256 + threadIdx.x;
    if (e >= E) return;
    float vx = vectors[e * 3 + 0], vy = vectors[e * 3 + 1], vz = vectors[e * 3 + 2];
    float r = sqrtf(vx * vx + vy * vy + vz * vz);
    float inv = 1.0f / r;
    float4 y; y.x = 1.0f; y.y = SQRT3 * vx * inv; y.z = SQRT3 * vy * inv; y.w = SQRT3 * vz * inv;
    *reinterpret_cast<float4*>(Yv + e * 4) = y;
    float f = 0.0f;
    if (r < 1.0f) {
        float u3 = r * r * r;
        float u6 = u3 * u3;
        f = 1.0f + u6 * (-28.0f + r * (48.0f - 21.0f * r));   // p = 6 envelope
    }
    env[e] = f;
}

__global__ __launch_bounds__(256) void zero_f4(float* __restrict__ p, long n4) {
    long i = (long)blockIdx.x * 256 + threadIdx.x;
    if (i >= n4) return;
    float4 z; z.x = z.y = z.z = z.w = 0.0f;
    *reinterpret_cast<float4*>(p + i * 4) = z;
}

// ---------------------------------------------------------------------------
// w = (x @ W_w)/sqrt(512) fused with segment-sum scatter of w*Y into agg.
// N=64; 8 waves = 2 row-bands x 4 col-tiles -> 32 edges per block.
// ---------------------------------------------------------------------------
__global__ __launch_bounds__(256) void gemm_w_scatter(
    const unsigned short* __restrict__ A, int lda,
    const unsigned short* __restrict__ Bp,
    const int* __restrict__ senders,
    const float* __restrict__ Yv,
    float* __restrict__ agg, float scale) {
    constexpr int K = 512, colTiles = 4;       // N = 64
    const int wave = threadIdx.x >> 5;
    const int lane = threadIdx.x & 31;
    const int laneLo = lane & 15, laneHi = lane >> 4;
    const int band = wave >> 2;                // 0..1
    const int wcol = wave & 3;                 // 0..3
    const long row0 = ((long)blockIdx.x * 2 + band) * 16;
    const int col0 = wcol * 16;

    v8f acc = {0.f, 0.f, 0.f, 0.f, 0.f, 0.f, 0.f, 0.f};
    const unsigned short* Arow = A + (row0 + laneLo) * (long)lda;
    for (int kb = 0; kb < K; kb += 32) {
        v16bf a = load_a_frag(Arow, kb, laneHi);
        v16bf b = load_b_frag(Bp + (((long)(kb >> 5) * colTiles + wcol) << 9) + (lane << 4));
        acc = wmma_bf16(a, b, acc);
    }
    const int m = col0 + laneLo;
    #pragma unroll
    for (int v = 0; v < 8; ++v) {
        long r = row0 + v + 8 * laneHi;
        int s = senders[r];
        float wv = acc[v] * scale;
        const float4 y = *reinterpret_cast<const float4*>(Yv + r * 4);
        float* base = agg + (((long)s * 64 + m) << 2);
        __hip_atomic_fetch_add(base + 0, wv * y.x, __ATOMIC_RELAXED, __HIP_MEMORY_SCOPE_AGENT);
        __hip_atomic_fetch_add(base + 1, wv * y.y, __ATOMIC_RELAXED, __HIP_MEMORY_SCOPE_AGENT);
        __hip_atomic_fetch_add(base + 2, wv * y.z, __ATOMIC_RELAXED, __HIP_MEMORY_SCOPE_AGENT);
        __hip_atomic_fetch_add(base + 3, wv * y.w, __ATOMIC_RELAXED, __HIP_MEMORY_SCOPE_AGENT);
    }
}

// ---------------------------------------------------------------------------
// Per-(edge, mul) tensor products: scal -> xcb[:,512:640], vecs -> (3E x 192) bf16
// ---------------------------------------------------------------------------
__global__ __launch_bounds__(256) void edge_products(
    const float* __restrict__ agg, const float* __restrict__ V,
    const int* __restrict__ senders,
    unsigned short* __restrict__ xcb, unsigned short* __restrict__ vecsb, long E) {
    long gid = (long)blockIdx.x * 256 + threadIdx.x;
    long e = gid >> 6;
    int m = (int)(gid & 63);
    if (e >= E) return;
    int s = senders[e];
    const float4 ag = *reinterpret_cast<const float4*>(agg + (((long)s * 64 + m) << 2));
    const float4 bv = *reinterpret_cast<const float4*>(V + ((e * 64 + m) << 2));
    float a0 = ag.x * EPSILON, a1x = ag.y * EPSILON, a1y = ag.z * EPSILON, a1z = ag.w * EPSILON;
    float b0 = bv.x, b1x = bv.y, b1y = bv.z, b1z = bv.w;

    xcb[e * 640 + 512 + m] = f2bf(a0 * b0);
    xcb[e * 640 + 576 + m] = f2bf((a1x * b1x + a1y * b1y + a1z * b1z) * INV_SQRT3);

    float cx = a1y * b1z - a1z * b1y;
    float cy = a1z * b1x - a1x * b1z;
    float cz = a1x * b1y - a1y * b1x;

    long base = (3 * e) * 192;
    vecsb[base + m]        = f2bf(a0 * b1x);
    vecsb[base + 64 + m]   = f2bf(a1x * b0);
    vecsb[base + 128 + m]  = f2bf(cx * INV_SQRT2);
    base += 192;
    vecsb[base + m]        = f2bf(a0 * b1y);
    vecsb[base + 64 + m]   = f2bf(a1y * b0);
    vecsb[base + 128 + m]  = f2bf(cy * INV_SQRT2);
    base += 192;
    vecsb[base + m]        = f2bf(a0 * b1z);
    vecsb[base + 64 + m]   = f2bf(a1z * b0);
    vecsb[base + 128 + m]  = f2bf(cz * INV_SQRT2);
}

// ---------------------------------------------------------------------------
// Generic WMMA GEMM with register-level B reuse.
//   CT = col tiles per wave, RB = row bands per wave.
//   waves-per-band WPB = (N/16)/CT ; wave bands WB = 8/WPB.
//   Block covers 16*WB*RB rows x N cols.
//   MODE: 0 = SiLU -> bf16, 1 = *env -> f32, 2 = V_out remap -> f32.
// ---------------------------------------------------------------------------
enum { OUT_SILU_BF16 = 0, OUT_ENV_F32 = 1, OUT_VOUT_F32 = 2 };

template <int K, int N, int CT, int RB, int MODE>
__global__ __launch_bounds__(256) void gemm_wmma(
    const unsigned short* __restrict__ A, int lda,
    const unsigned short* __restrict__ Bp,
    unsigned short* __restrict__ outB,
    float* __restrict__ outF,
    const float* __restrict__ env, float scale) {
    constexpr int colTiles = N / 16;
    constexpr int WPB = colTiles / CT;         // waves spanning the N dimension
    constexpr int WB  = 8 / WPB;               // wave-level row bands
    static_assert(colTiles % CT == 0 && 8 % WPB == 0, "tiling");

    const int wave = threadIdx.x >> 5;
    const int lane = threadIdx.x & 31;
    const int laneLo = lane & 15, laneHi = lane >> 4;
    const int waveBand = wave / WPB;           // 0..WB-1
    const int wcol = wave % WPB;
    const int col0 = wcol * CT * 16;
    const long bandBase = ((long)blockIdx.x * WB + waveBand) * RB;  // in 16-row units

    v8f acc[RB][CT];
    #pragma unroll
    for (int b = 0; b < RB; ++b)
        #pragma unroll
        for (int t = 0; t < CT; ++t)
            acc[b][t] = v8f{0.f, 0.f, 0.f, 0.f, 0.f, 0.f, 0.f, 0.f};

    const unsigned short* Arow[RB];
    #pragma unroll
    for (int b = 0; b < RB; ++b)
        Arow[b] = A + ((bandBase + b) * 16 + laneLo) * (long)lda;

    for (int kb = 0; kb < K; kb += 32) {
        v16bf a[RB];
        #pragma unroll
        for (int b = 0; b < RB; ++b) a[b] = load_a_frag(Arow[b], kb, laneHi);
        const long kt = kb >> 5;
        #pragma unroll
        for (int t = 0; t < CT; ++t) {
            const long nt = (col0 >> 4) + t;
            v16bf bf = load_b_frag(Bp + ((kt * colTiles + nt) << 9) + (lane << 4));
            #pragma unroll
            for (int b = 0; b < RB; ++b)
                acc[b][t] = wmma_bf16(a[b], bf, acc[b][t]);
        }
    }
    #pragma unroll
    for (int b = 0; b < RB; ++b) {
        const long row0 = (bandBase + b) * 16;
        #pragma unroll
        for (int t = 0; t < CT; ++t) {
            #pragma unroll
            for (int v = 0; v < 8; ++v) {
                long r = row0 + v + 8 * laneHi;
                int c = col0 + t * 16 + laneLo;
                float z = acc[b][t][v] * scale;
                if constexpr (MODE == OUT_SILU_BF16) {
                    z = z / (1.0f + __expf(-z));          // SiLU
                    outB[r * N + c] = f2bf(z);
                } else if constexpr (MODE == OUT_ENV_F32) {
                    outF[r * N + c] = z * env[r];
                } else {                                   // V_out: row = 3e+c_comp
                    long e = r / 3;
                    int cc = (int)(r - e * 3);
                    outF[e * 192 + (long)c * 3 + cc] = z;
                }
            }
        }
    }
}

// ---------------------------------------------------------------------------
// Host launcher
// ---------------------------------------------------------------------------
extern "C" void kernel_launch(void* const* d_in, const int* in_sizes, int n_in,
                              void* d_out, int out_size, void* d_ws, size_t ws_size,
                              hipStream_t stream) {
    const float* vectors = (const float*)d_in[0];
    const float* x       = (const float*)d_in[1];
    const float* V       = (const float*)d_in[2];
    const int*   senders = (const int*)d_in[3];
    const float* W_w     = (const float*)d_in[4];
    const float* W1      = (const float*)d_in[5];
    const float* W2      = (const float*)d_in[6];
    const float* W3      = (const float*)d_in[7];
    const float* W_lin   = (const float*)d_in[8];
    float* out = (float*)d_out;

    const long E = in_sizes[0] / 3;           // 131072
    const long NODES = 8192;

    // ---- workspace partition (256B aligned) ----
    char* ws = (char*)d_ws;
    size_t off = 0;
    auto alloc = [&](size_t bytes) { char* p = ws + off; off = (off + bytes + 255) & ~(size_t)255; return p; };
    unsigned short* xcb    = (unsigned short*)alloc((size_t)E * 640 * 2);  // xc bf16 (reused as h2)
    unsigned short* h1b    = (unsigned short*)alloc((size_t)E * 512 * 2);
    unsigned short* vecsb  = (unsigned short*)alloc((size_t)3 * E * 192 * 2);
    float*          agg    = (float*)alloc((size_t)NODES * 64 * 4 * 4);
    float*          Yv     = (float*)alloc((size_t)E * 4 * 4);
    float*          env    = (float*)alloc((size_t)E * 4);
    unsigned short* Wwp    = (unsigned short*)alloc((size_t)512 * 64 * 2);
    unsigned short* W1p    = (unsigned short*)alloc((size_t)640 * 512 * 2);
    unsigned short* W2p    = (unsigned short*)alloc((size_t)512 * 512 * 2);
    unsigned short* W3p    = (unsigned short*)alloc((size_t)512 * 512 * 2);
    unsigned short* Wlinp  = (unsigned short*)alloc((size_t)192 * 64 * 2);
    unsigned short* h2b    = xcb;   // xc dead after GEMM1 -> reuse for h2

    const float sW  = 1.0f / sqrtf(512.0f);
    const float s1  = 1.0f / sqrtf(640.0f);
    const float s2  = 1.0f / sqrtf(512.0f);
    const float sV  = 1.0f / sqrtf(192.0f);

    // 1. pack weights into fragment-major bf16
    auto packGrid = [](int K, int N) { return ((K / 32) * (N / 16) * 32 + 255) / 256; };
    pack_weights<<<packGrid(512, 64),  256, 0, stream>>>(W_w,   Wwp,   512, 64);
    pack_weights<<<packGrid(640, 512), 256, 0, stream>>>(W1,    W1p,   640, 512);
    pack_weights<<<packGrid(512, 512), 256, 0, stream>>>(W2,    W2p,   512, 512);
    pack_weights<<<packGrid(512, 512), 256, 0, stream>>>(W3,    W3p,   512, 512);
    pack_weights<<<packGrid(192, 64),  256, 0, stream>>>(W_lin, Wlinp, 192, 64);

    // 2. x -> bf16 into xc[:, :512]
    xcopy<<<(unsigned)(E * 128 / 256), 256, 0, stream>>>(x, xcb, E * 128);

    // 3. geometry
    geom<<<(unsigned)((E + 255) / 256), 256, 0, stream>>>(vectors, Yv, env, E);

    // 4. zero agg, then w-GEMM + scatter (segment_sum)
    const long aggN4 = NODES * 64;    // float4 count
    zero_f4<<<(unsigned)((aggN4 + 255) / 256), 256, 0, stream>>>(agg, aggN4);
    gemm_w_scatter<<<(unsigned)(E / 32), 256, 0, stream>>>(xcb, 640, Wwp, senders, Yv, agg, sW);

    // 5. edge tensor products
    edge_products<<<(unsigned)(E * 64 / 256), 256, 0, stream>>>(agg, V, senders, xcb, vecsb, E);

    // 6. MLP: h1 = silu(xc@W1/s), h2 = silu(h1@W2/s), x_out = env * (h2@W3/s)
    //    RB=2: each B fragment feeds two row-band WMMAs (halves B traffic).
    gemm_wmma<640, 512, 4, 2, OUT_SILU_BF16><<<(unsigned)(E / 32), 256, 0, stream>>>(
        xcb, 640, W1p, h1b, nullptr, nullptr, s1);
    gemm_wmma<512, 512, 4, 2, OUT_SILU_BF16><<<(unsigned)(E / 32), 256, 0, stream>>>(
        h1b, 512, W2p, h2b, nullptr, nullptr, s2);
    gemm_wmma<512, 512, 4, 2, OUT_ENV_F32><<<(unsigned)(E / 32), 256, 0, stream>>>(
        h2b, 512, W3p, nullptr, out, env, s2);

    // 7. V_out = (vecs @ W_lin)/sqrt(192), rows = 3E, remapped to (E,64,3)
    gemm_wmma<192, 64, 1, 1, OUT_VOUT_F32><<<(unsigned)(3 * E / 32), 256, 0, stream>>>(
        vecsb, 192, Wlinp, nullptr, out + (size_t)E * 512, nullptr, sV);

    (void)n_in; (void)out_size; (void)ws_size;
}
